// GIN_18657337933844
// MI455X (gfx1250) — compile-verified, matrix operand order checked by reading, not run on hardware
//
#include <hip/hip_runtime.h>
#include <hip/hip_bf16.h>
#include <math.h>

#define HID     128
#define NGRAPH  64
#define NCLS    10
#define WSTRIDE 136   // ushort pitch for LDS weight tiles [n][k] (272B rows: 16B-aligned, bank-step 4)
#define SSTRIDE 136   // ushort pitch for per-wave activation stage [m][k]

typedef __attribute__((ext_vector_type(16))) __bf16 v16bf;
typedef __attribute__((ext_vector_type(8)))  float  v8f;

union Frag { v16bf v; uint4 q[2]; };

__device__ __forceinline__ unsigned short f2bf(float f) {
    unsigned int u = __float_as_uint(f);
    unsigned int r = u + 0x7FFFu + ((u >> 16) & 1u);   // round-to-nearest-even
    return (unsigned short)(r >> 16);
}

// ---------------------------------------------------------------------------
// z_pre[i] = (1 + eps[l]) * h[i]
// ---------------------------------------------------------------------------
__global__ void gin_preagg_kernel(const float4* __restrict__ h4,
                                  const float* __restrict__ eps, int l,
                                  float4* __restrict__ z4, int total4) {
    int i = blockIdx.x * blockDim.x + threadIdx.x;
    if (i >= total4) return;
    float s = 1.0f + eps[l];
    float4 f = h4[i];
    z4[i] = make_float4(s * f.x, s * f.y, s * f.z, s * f.w);
}

// ---------------------------------------------------------------------------
// One wave32 per edge: z_pre[dst] += h[src] (lane handles 4 consecutive feats)
// ---------------------------------------------------------------------------
__global__ __launch_bounds__(256) void gin_scatter_kernel(
        const float* __restrict__ h, const int* __restrict__ src,
        const int* __restrict__ dst, float* __restrict__ zpre, int ne) {
    int e    = (blockIdx.x * blockDim.x + threadIdx.x) >> 5;
    int lane = threadIdx.x & 31;
    if (e >= ne) return;
    int s = src[e], d = dst[e];
    const float4 f = *(const float4*)(&h[(size_t)s * HID + lane * 4]);
    float* p = &zpre[(size_t)d * HID + lane * 4];
    atomicAdd(p + 0, f.x);
    atomicAdd(p + 1, f.y);
    atomicAdd(p + 2, f.z);
    atomicAdd(p + 3, f.w);
}

// ---------------------------------------------------------------------------
// Fused GIN MLP: h_out = relu(BN(relu(relu(z@W1+b1)@W2+b2)))
// 256 threads = 8 waves; each wave owns 16 rows; WG covers 128 rows.
// W1/W2 live in LDS as bf16 transposed [n][k]; per-wave bf16 stage tile
// carries activations between the two WMMA GEMMs.
// ---------------------------------------------------------------------------
__global__ __launch_bounds__(256) void gin_mlp_kernel(
        const float* __restrict__ zpre,
        const float* __restrict__ W1, const float* __restrict__ b1,
        const float* __restrict__ W2, const float* __restrict__ b2,
        const float* __restrict__ gamma, const float* __restrict__ beta,
        const float* __restrict__ rmean, const float* __restrict__ rvar,
        float* __restrict__ hout, int nrows) {
    __shared__ unsigned short lw1[HID * WSTRIDE];
    __shared__ unsigned short lw2[HID * WSTRIDE];
    __shared__ unsigned short stage[8 * 16 * SSTRIDE];

    const int tid  = threadIdx.x;
    const int wave = tid >> 5;
    const int lane = tid & 31;
    const int half = lane >> 4;    // 0: K-low group, 1: K-high group
    const int l16  = lane & 15;

    // Load + transpose weights to LDS (bf16, column-major [n][k]).
    for (int i = tid; i < HID * HID; i += 256) {
        int k = i >> 7, n = i & 127;
        lw1[n * WSTRIDE + k] = f2bf(W1[i]);
        lw2[n * WSTRIDE + k] = f2bf(W2[i]);
    }
    __syncthreads();

    const int row0 = blockIdx.x * 128 + wave * 16;
    unsigned short* st = &stage[wave * 16 * SSTRIDE];

    // Stage this wave's 16 rows of z_pre as bf16 (rows past N clamp to last row).
    for (int i = 0; i < 16; ++i) {
        int r  = row0 + i;
        int rr = r < nrows ? r : (nrows - 1);
        const float4 f = *(const float4*)(&zpre[(size_t)rr * HID + lane * 4]);
        unsigned int p0 = (unsigned)f2bf(f.x) | ((unsigned)f2bf(f.y) << 16);
        unsigned int p1 = (unsigned)f2bf(f.z) | ((unsigned)f2bf(f.w) << 16);
        *(uint2*)(&st[i * SSTRIDE + lane * 4]) = make_uint2(p0, p1);
    }

    // ISA §7.12.2 16-bit fragment: elems 0..7 <- K[kb..kb+7], 8..15 <- K[kb+16..kb+23],
    // kb = s*32 + half*8. Two aligned b128 LDS loads per fragment.
    auto loadfrag = [&](const unsigned short* base, int pitch, int row, int kb) {
        Frag f;
        f.q[0] = *(const uint4*)(&base[row * pitch + kb]);
        f.q[1] = *(const uint4*)(&base[row * pitch + kb + 16]);
        return f;
    };

    // A fragments for GEMM1 (4 K-steps of 32).
    Frag a1[4];
    #pragma unroll
    for (int s = 0; s < 4; ++s)
        a1[s] = loadfrag(st, SSTRIDE, l16, s * 32 + half * 8);

    // GEMM1: z1 = relu(z @ W1 + b1) -> bf16 back into the wave's stage tile.
    #pragma unroll
    for (int n = 0; n < 8; ++n) {
        v8f acc = {};
        #pragma unroll
        for (int s = 0; s < 4; ++s) {
            Frag b = loadfrag(lw1, WSTRIDE, n * 16 + l16, s * 32 + half * 8);
            acc = __builtin_amdgcn_wmma_f32_16x16x32_bf16(
                      false, a1[s].v, false, b.v, (short)0, acc, false, false);
        }
        int col = n * 16 + l16;
        float bias = b1[col];
        #pragma unroll
        for (int r = 0; r < 8; ++r) {
            float v = acc[r] + bias;
            v = v > 0.0f ? v : 0.0f;
            st[(half * 8 + r) * SSTRIDE + col] = f2bf(v);   // D: VGPR r -> row half*8+r
        }
    }

    // A fragments for GEMM2 (from z1 in stage; same-wave LDS is in-order).
    Frag a2[4];
    #pragma unroll
    for (int s = 0; s < 4; ++s)
        a2[s] = loadfrag(st, SSTRIDE, l16, s * 32 + half * 8);

    // GEMM2 + bias + relu + BN(eval) + outer relu -> global f32.
    #pragma unroll
    for (int n = 0; n < 8; ++n) {
        v8f acc = {};
        #pragma unroll
        for (int s = 0; s < 4; ++s) {
            Frag b = loadfrag(lw2, WSTRIDE, n * 16 + l16, s * 32 + half * 8);
            acc = __builtin_amdgcn_wmma_f32_16x16x32_bf16(
                      false, a2[s].v, false, b.v, (short)0, acc, false, false);
        }
        int col  = n * 16 + l16;
        float bias = b2[col];
        float sc = gamma[col] * rsqrtf(rvar[col] + 1e-5f);
        float sh = beta[col] - rmean[col] * sc;
        #pragma unroll
        for (int r = 0; r < 8; ++r) {
            float v = acc[r] + bias;
            v = v > 0.0f ? v : 0.0f;     // MLP relu
            v = fmaf(v, sc, sh);         // BatchNorm (running stats)
            v = v > 0.0f ? v : 0.0f;     // outer relu
            int m = row0 + half * 8 + r;
            if (m < nrows) hout[(size_t)m * HID + col] = v;
        }
    }
}

// ---------------------------------------------------------------------------
// global_add_pool
// ---------------------------------------------------------------------------
__global__ void pool_zero_kernel(float* __restrict__ pooled) {
    int i = blockIdx.x * blockDim.x + threadIdx.x;
    if (i < NGRAPH * HID) pooled[i] = 0.0f;
}

__global__ __launch_bounds__(256) void pool_scatter_kernel(
        const float* __restrict__ h, const int* __restrict__ batch,
        float* __restrict__ pooled, int n) {
    int i    = (blockIdx.x * blockDim.x + threadIdx.x) >> 5;
    int lane = threadIdx.x & 31;
    if (i >= n) return;
    int g = batch[i];
    const float4 f = *(const float4*)(&h[(size_t)i * HID + lane * 4]);
    float* p = &pooled[g * HID + lane * 4];
    atomicAdd(p + 0, f.x);
    atomicAdd(p + 1, f.y);
    atomicAdd(p + 2, f.z);
    atomicAdd(p + 3, f.w);
}

// ---------------------------------------------------------------------------
// Head MLP + log_softmax (tiny: one workgroup)
// ---------------------------------------------------------------------------
__global__ __launch_bounds__(128) void gin_head_kernel(
        const float* __restrict__ pooled,
        const float* __restrict__ fc1w, const float* __restrict__ fc1b,
        const float* __restrict__ fc2w, const float* __restrict__ fc2b,
        const float* __restrict__ fc3w, const float* __restrict__ fc3b,
        float* __restrict__ out) {
    __shared__ float Y1[NGRAPH][HID];
    __shared__ float Y2[NGRAPH][HID / 2];
    __shared__ float S[NGRAPH][NCLS];
    int t = threadIdx.x;

    for (int g = 0; g < NGRAPH; ++g) {
        float acc = fc1b[t];
        for (int k = 0; k < HID; ++k) acc += pooled[g * HID + k] * fc1w[k * HID + t];
        Y1[g][t] = acc > 0.0f ? acc : 0.0f;
    }
    __syncthreads();
    if (t < HID / 2) {
        for (int g = 0; g < NGRAPH; ++g) {
            float acc = fc2b[t];
            for (int k = 0; k < HID; ++k) acc += Y1[g][k] * fc2w[k * (HID / 2) + t];
            Y2[g][t] = acc > 0.0f ? acc : 0.0f;
        }
    }
    __syncthreads();
    if (t < NCLS) {
        for (int g = 0; g < NGRAPH; ++g) {
            float acc = fc3b[t];
            for (int k = 0; k < HID / 2; ++k) acc += Y2[g][k] * fc3w[k * NCLS + t];
            S[g][t] = acc;
        }
    }
    __syncthreads();
    if (t < NGRAPH) {
        float mx = S[t][0];
        for (int c = 1; c < NCLS; ++c) mx = fmaxf(mx, S[t][c]);
        float se = 0.0f;
        for (int c = 0; c < NCLS; ++c) se += __expf(S[t][c] - mx);
        float lse = mx + __logf(se);
        for (int c = 0; c < NCLS; ++c) out[t * NCLS + c] = S[t][c] - lse;
    }
}

// ---------------------------------------------------------------------------
extern "C" void kernel_launch(void* const* d_in, const int* in_sizes, int n_in,
                              void* d_out, int out_size, void* d_ws, size_t ws_size,
                              hipStream_t stream) {
    const float* x     = (const float*)d_in[0];
    const int*   eidx  = (const int*)d_in[1];
    const int*   batch = (const int*)d_in[2];
    const float* eps   = (const float*)d_in[3];
    const float* W1    = (const float*)d_in[4];
    const float* b1    = (const float*)d_in[5];
    const float* W2    = (const float*)d_in[6];
    const float* b2    = (const float*)d_in[7];
    const float* gamma = (const float*)d_in[8];
    const float* beta  = (const float*)d_in[9];
    const float* rm    = (const float*)d_in[10];
    const float* rv    = (const float*)d_in[11];
    const float* fc1w  = (const float*)d_in[12];
    const float* fc1b  = (const float*)d_in[13];
    const float* fc2w  = (const float*)d_in[14];
    const float* fc2b  = (const float*)d_in[15];
    const float* fc3w  = (const float*)d_in[16];
    const float* fc3b  = (const float*)d_in[17];
    float* out = (float*)d_out;

    const int N = in_sizes[0] / HID;        // 100000
    const int E = in_sizes[1] / 2;          // 1600000

    float* zpre   = (float*)d_ws;                    // N*128 f32
    float* hbuf   = zpre + (size_t)N * HID;          // N*128 f32
    float* pooled = hbuf + (size_t)N * HID;          // 64*128 f32

    const int* srcp = eidx;
    const int* dstp = eidx + E;

    const float* hcur = x;
    for (int l = 0; l < 3; ++l) {
        gin_preagg_kernel<<<(N * (HID / 4) + 255) / 256, 256, 0, stream>>>(
            (const float4*)hcur, eps, l, (float4*)zpre, N * (HID / 4));
        gin_scatter_kernel<<<(E + 7) / 8, 256, 0, stream>>>(hcur, srcp, dstp, zpre, E);
        gin_mlp_kernel<<<(N + 127) / 128, 256, 0, stream>>>(
            zpre,
            W1 + (size_t)l * HID * HID, b1 + l * HID,
            W2 + (size_t)l * HID * HID, b2 + l * HID,
            gamma + l * HID, beta + l * HID, rm + l * HID, rv + l * HID,
            hbuf, N);
        hcur = hbuf;
    }
    pool_zero_kernel<<<(NGRAPH * HID + 255) / 256, 256, 0, stream>>>(pooled);
    pool_scatter_kernel<<<(N + 7) / 8, 256, 0, stream>>>(hcur, batch, pooled, N);
    gin_head_kernel<<<1, 128, 0, stream>>>(pooled, fc1w, fc1b, fc2w, fc2b, fc3w, fc3b, out);
}
